// Qwen3MoEMLP_43885975830869
// MI455X (gfx1250) — compile-verified, matrix operand order checked by reading, not run on hardware
//
#include <hip/hip_runtime.h>
#include <hip/hip_bf16.h>
#include <stdint.h>

#define BT    4096     // B*T tokens
#define EDIM  1024
#define INTER 1408
#define NEXP  8
#define LDA   40       // padded LDS row stride (bf16) for 32-wide K tiles
#define LDB2  72       // padded LDS row stride (bf16) for 64-wide K tiles

typedef __attribute__((ext_vector_type(16))) __bf16 v16bf;
typedef __attribute__((ext_vector_type(8)))  __bf16 v8bf;
typedef __attribute__((ext_vector_type(8)))  float  v8f;
typedef __attribute__((ext_vector_type(4)))  int    v4i;

// ---- gfx1250 async global->LDS support (guarded probe; falls back to VGPR staging) ----
#if defined(__has_builtin)
#  if __has_builtin(__builtin_amdgcn_global_load_async_to_lds_b128)
#    define USE_ASYNC_LDS 1
#  endif
#endif
#ifndef USE_ASYNC_LDS
#  define USE_ASYNC_LDS 0
#endif

#if USE_ASYNC_LDS
#  if __has_builtin(__builtin_amdgcn_s_wait_asynccnt)
#    define WAIT_ASYNC0() __builtin_amdgcn_s_wait_asynccnt(0)
#  else
#    define WAIT_ASYNC0() asm volatile("s_wait_asynccnt 0x0" ::: "memory")
#  endif
typedef __attribute__((address_space(1))) v4i* gas_v4i_p;  // global (device) AS
typedef __attribute__((address_space(3))) v4i* las_v4i_p;  // LDS AS
__device__ __forceinline__ void async_copy16(const unsigned short* g, unsigned short* l) {
  // generic->AS1 is numerically identity; generic LDS low 32 bits are the LDS byte address
  __builtin_amdgcn_global_load_async_to_lds_b128(
      (gas_v4i_p)(size_t)g, (las_v4i_p)(unsigned int)(size_t)l, 0, 0);
}
#endif

__device__ __forceinline__ unsigned short f32_to_bf16(float f) {
  unsigned int u = __float_as_uint(f);
  u += 0x7FFFu + ((u >> 16) & 1u);           // round to nearest even
  return (unsigned short)(u >> 16);
}

__device__ __forceinline__ v16bf load_frag(const unsigned short* p) {
  // Lane layout per ISA 7.12.2 (16-bit A/B 16x32): lane(lo) K 0..7 & 16..23,
  // lane(hi) K 8..15 & 24..31; caller passes p already offset by lh*8.
  v8bf lo = *(const v8bf*)p;
  v8bf hi = *(const v8bf*)(p + 16);
  return __builtin_shufflevector(lo, hi, 0,1,2,3,4,5,6,7,8,9,10,11,12,13,14,15);
}

// ---------------- fp32 -> bf16 convert ----------------
__global__ __launch_bounds__(256)
void convert_f32_bf16(const float* __restrict__ in, unsigned short* __restrict__ out, int n) {
  int i = blockIdx.x * blockDim.x + threadIdx.x;
  int stride = gridDim.x * blockDim.x;
  for (; i < n; i += stride) out[i] = f32_to_bf16(in[i]);
}

// ---------------- router: top-2 softmax gating ----------------
__global__ __launch_bounds__(256)
void moe_router(const float* __restrict__ x, const float* __restrict__ gw,
                float* __restrict__ gate) {
  __shared__ float sgw[NEXP * EDIM];   // 32 KB
  const int tid = threadIdx.x;
  for (int i = tid; i < NEXP * EDIM; i += 256) sgw[i] = gw[i];
  __syncthreads();

  const int t = blockIdx.x * 256 + tid;
  const float* xr = x + (size_t)t * EDIM;
  float acc[NEXP];
#pragma unroll
  for (int e = 0; e < NEXP; ++e) acc[e] = 0.f;
  for (int j = 0; j < EDIM; ++j) {
    float xv = xr[j];
#pragma unroll
    for (int e = 0; e < NEXP; ++e) acc[e] = fmaf(xv, sgw[e * EDIM + j], acc[e]);
  }
  float s0 = acc[0]; int i0 = 0;
#pragma unroll
  for (int e = 1; e < NEXP; ++e) if (acc[e] > s0) { s0 = acc[e]; i0 = e; }
  float s1 = -3.4e38f; int i1 = 0;
#pragma unroll
  for (int e = 0; e < NEXP; ++e) if (e != i0 && acc[e] > s1) { s1 = acc[e]; i1 = e; }
  float p1 = __expf(s1 - s0);
  float inv = 1.f / (1.f + p1);
#pragma unroll
  for (int e = 0; e < NEXP; ++e) {
    float gv = (e == i0) ? inv : ((e == i1) ? p1 * inv : 0.f);
    gate[(size_t)t * NEXP + e] = gv;
  }
}

// ---------------- GEMM1: H' = gate * silu(X Wg^T) * (X Wu^T), bf16 out ----------------
// Block tile 128(M) x 64(N), K-stage 32, 8 waves (4x2), wave tile 32x32.
__global__ __launch_bounds__(256)
void moe_gemm1(const unsigned short* __restrict__ xbf,
               const unsigned short* __restrict__ wgbf,
               const unsigned short* __restrict__ wubf,
               const float* __restrict__ gate,
               unsigned short* __restrict__ hp) {
  __shared__ unsigned short sA[2][128 * LDA];
  __shared__ unsigned short sG[2][64 * LDA];
  __shared__ unsigned short sU[2][64 * LDA];
  __shared__ float sgate[128];

  const int tid   = threadIdx.x;
  const int e     = blockIdx.z;
  const int tileM = blockIdx.y * 128;
  const int tileN = blockIdx.x * 64;

  if (tid < 128) sgate[tid] = gate[(size_t)(tileM + tid) * NEXP + e];

  const int lane = tid & 31, lr = lane & 15, lh = lane >> 4;
  const int w = tid >> 5, wm = w >> 1, wn = w & 1;

  const int arow = tid >> 2, ac = (tid & 3) * 8;
  const int brow = tid >> 2, bc = (tid & 3) * 8;

  const unsigned short* gAbase = xbf + (size_t)tileM * EDIM;
  const unsigned short* gGbase = wgbf + ((size_t)e * INTER + tileN) * EDIM;
  const unsigned short* gUbase = wubf + ((size_t)e * INTER + tileN) * EDIM;

#if USE_ASYNC_LDS
  auto issueStage = [&](int k0, int buf) {
    async_copy16(gAbase + (size_t)arow * EDIM + k0 + ac,        &sA[buf][arow * LDA + ac]);
    async_copy16(gAbase + (size_t)(arow + 64) * EDIM + k0 + ac, &sA[buf][(arow + 64) * LDA + ac]);
    async_copy16(gGbase + (size_t)brow * EDIM + k0 + bc,        &sG[buf][brow * LDA + bc]);
    async_copy16(gUbase + (size_t)brow * EDIM + k0 + bc,        &sU[buf][brow * LDA + bc]);
  };
#else
  uint4 rA0, rA1, rG, rU;
  auto loadStage = [&](int k0) {
    rA0 = *(const uint4*)(gAbase + (size_t)arow * EDIM + k0 + ac);
    rA1 = *(const uint4*)(gAbase + (size_t)(arow + 64) * EDIM + k0 + ac);
    rG  = *(const uint4*)(gGbase + (size_t)brow * EDIM + k0 + bc);
    rU  = *(const uint4*)(gUbase + (size_t)brow * EDIM + k0 + bc);
  };
  auto storeStage = [&](int buf) {
    *(uint4*)&sA[buf][arow * LDA + ac]        = rA0;
    *(uint4*)&sA[buf][(arow + 64) * LDA + ac] = rA1;
    *(uint4*)&sG[buf][brow * LDA + bc]        = rG;
    *(uint4*)&sU[buf][brow * LDA + bc]        = rU;
  };
#endif

  v8f accg[2][2] = {{{0.f},{0.f}},{{0.f},{0.f}}};
  v8f accu[2][2] = {{{0.f},{0.f}},{{0.f},{0.f}}};

#if USE_ASYNC_LDS
  issueStage(0, 0);
  WAIT_ASYNC0();
#else
  loadStage(0);
  storeStage(0);
#endif
  __syncthreads();

  const int KIT = EDIM / 32;
  for (int kk = 0; kk < KIT; ++kk) {
    const int cur = kk & 1;
#if USE_ASYNC_LDS
    if (kk + 1 < KIT) issueStage((kk + 1) * 32, cur ^ 1);
#else
    if (kk + 1 < KIT) loadStage((kk + 1) * 32);
#endif
    if (kk + 2 < KIT) {                      // gfx1250 global_prefetch_b8 path
      __builtin_prefetch(gAbase + (size_t)arow * EDIM + (kk + 2) * 32 + ac, 0, 1);
      __builtin_prefetch(gGbase + (size_t)brow * EDIM + (kk + 2) * 32 + bc, 0, 1);
      __builtin_prefetch(gUbase + (size_t)brow * EDIM + (kk + 2) * 32 + bc, 0, 1);
    }
    v16bf a[2], bg[2], bu[2];
#pragma unroll
    for (int mf = 0; mf < 2; ++mf)
      a[mf] = load_frag(&sA[cur][(wm * 32 + mf * 16 + lr) * LDA + lh * 8]);
#pragma unroll
    for (int nf = 0; nf < 2; ++nf) {
      bg[nf] = load_frag(&sG[cur][(wn * 32 + nf * 16 + lr) * LDA + lh * 8]);
      bu[nf] = load_frag(&sU[cur][(wn * 32 + nf * 16 + lr) * LDA + lh * 8]);
    }
#pragma unroll
    for (int mf = 0; mf < 2; ++mf)
#pragma unroll
      for (int nf = 0; nf < 2; ++nf) {
        accg[mf][nf] = __builtin_amdgcn_wmma_f32_16x16x32_bf16(
            false, a[mf], false, bg[nf], (short)0, accg[mf][nf], false, false);
        accu[mf][nf] = __builtin_amdgcn_wmma_f32_16x16x32_bf16(
            false, a[mf], false, bu[nf], (short)0, accu[mf][nf], false, false);
      }
#if USE_ASYNC_LDS
    if (kk + 1 < KIT) WAIT_ASYNC0();
#else
    if (kk + 1 < KIT) storeStage(cur ^ 1);
#endif
    __syncthreads();
  }

  // Epilogue: silu(g)*u*gate -> bf16 H'
#pragma unroll
  for (int mf = 0; mf < 2; ++mf) {
#pragma unroll
    for (int nf = 0; nf < 2; ++nf) {
      const int ncol = tileN + wn * 32 + nf * 16 + lr;
#pragma unroll
      for (int r = 0; r < 8; ++r) {
        const int mloc = wm * 32 + mf * 16 + lh * 8 + r;
        float g = accg[mf][nf][r];
        float u = accu[mf][nf][r];
        float s = g / (1.f + __expf(-g));
        float h = s * u * sgate[mloc];
        hp[((size_t)e * BT + tileM + mloc) * INTER + ncol] = f32_to_bf16(h);
      }
    }
  }
}

// ---------------- GEMM2: Y = sum_e H'_e Wd_e^T (K = 8*1408), f32 out ----------------
// Block tile 128(M) x 64(N), K-stage 64 (2 WMMA slices/stage -> 176 barriers).
__global__ __launch_bounds__(256)
void moe_gemm2(const unsigned short* __restrict__ hp,
               const unsigned short* __restrict__ wdbf,
               float* __restrict__ y) {
  __shared__ unsigned short sA[2][128 * LDB2];   // 36 KB
  __shared__ unsigned short sB[2][64 * LDB2];    // 18 KB

  const int tid   = threadIdx.x;
  const int tileM = blockIdx.y * 128;
  const int tileN = blockIdx.x * 64;

  const int lane = tid & 31, lr = lane & 15, lh = lane >> 4;
  const int w = tid >> 5, wm = w >> 1, wn = w & 1;

  const int ar   = tid >> 3;         // 0..31
  const int ac8  = (tid & 7) * 8;    // col chunk within 64-wide stage

  const int IK  = INTER / 64;        // 22
  const int KIT = NEXP * IK;         // 176

  auto gbase = [&](int kk, const unsigned short*& pa, const unsigned short*& pb) {
    const int e  = kk / IK;
    const int i0 = (kk - e * IK) * 64;
    pa = hp   + ((size_t)e * BT   + tileM) * INTER + i0;
    pb = wdbf + ((size_t)e * EDIM + tileN) * INTER + i0;
  };

#if USE_ASYNC_LDS
  auto issueStage = [&](int kk, int buf) {
    const unsigned short *pa, *pb; gbase(kk, pa, pb);
#pragma unroll
    for (int q = 0; q < 4; ++q)
      async_copy16(pa + (size_t)(ar + 32 * q) * INTER + ac8, &sA[buf][(ar + 32 * q) * LDB2 + ac8]);
#pragma unroll
    for (int q = 0; q < 2; ++q)
      async_copy16(pb + (size_t)(ar + 32 * q) * INTER + ac8, &sB[buf][(ar + 32 * q) * LDB2 + ac8]);
  };
#else
  uint4 rA[4], rB[2];
  auto loadStage = [&](int kk) {
    const unsigned short *pa, *pb; gbase(kk, pa, pb);
#pragma unroll
    for (int q = 0; q < 4; ++q) rA[q] = *(const uint4*)(pa + (size_t)(ar + 32 * q) * INTER + ac8);
#pragma unroll
    for (int q = 0; q < 2; ++q) rB[q] = *(const uint4*)(pb + (size_t)(ar + 32 * q) * INTER + ac8);
  };
  auto storeStage = [&](int buf) {
#pragma unroll
    for (int q = 0; q < 4; ++q) *(uint4*)&sA[buf][(ar + 32 * q) * LDB2 + ac8] = rA[q];
#pragma unroll
    for (int q = 0; q < 2; ++q) *(uint4*)&sB[buf][(ar + 32 * q) * LDB2 + ac8] = rB[q];
  };
#endif

  v8f acc[2][2] = {{{0.f},{0.f}},{{0.f},{0.f}}};

#if USE_ASYNC_LDS
  issueStage(0, 0);
  WAIT_ASYNC0();
#else
  loadStage(0);
  storeStage(0);
#endif
  __syncthreads();

  for (int kk = 0; kk < KIT; ++kk) {
    const int cur = kk & 1;
#if USE_ASYNC_LDS
    if (kk + 1 < KIT) issueStage(kk + 1, cur ^ 1);
#else
    if (kk + 1 < KIT) loadStage(kk + 1);
#endif
    if (kk + 2 < KIT) {
      const unsigned short *pa, *pb; gbase(kk + 2, pa, pb);
      __builtin_prefetch(pa + (size_t)ar * INTER + ac8, 0, 1);
      __builtin_prefetch(pb + (size_t)ar * INTER + ac8, 0, 1);
    }
#pragma unroll
    for (int s = 0; s < 2; ++s) {          // two 32-wide WMMA K-slices per stage
      v16bf a[2], b[2];
#pragma unroll
      for (int mf = 0; mf < 2; ++mf)
        a[mf] = load_frag(&sA[cur][(wm * 32 + mf * 16 + lr) * LDB2 + s * 32 + lh * 8]);
#pragma unroll
      for (int nf = 0; nf < 2; ++nf)
        b[nf] = load_frag(&sB[cur][(wn * 32 + nf * 16 + lr) * LDB2 + s * 32 + lh * 8]);
#pragma unroll
      for (int mf = 0; mf < 2; ++mf)
#pragma unroll
        for (int nf = 0; nf < 2; ++nf)
          acc[mf][nf] = __builtin_amdgcn_wmma_f32_16x16x32_bf16(
              false, a[mf], false, b[nf], (short)0, acc[mf][nf], false, false);
    }
#if USE_ASYNC_LDS
    if (kk + 1 < KIT) WAIT_ASYNC0();
#else
    if (kk + 1 < KIT) storeStage(cur ^ 1);
#endif
    __syncthreads();
  }

#pragma unroll
  for (int mf = 0; mf < 2; ++mf) {
#pragma unroll
    for (int nf = 0; nf < 2; ++nf) {
      const int ncol = tileN + wn * 32 + nf * 16 + lr;
#pragma unroll
      for (int r = 0; r < 8; ++r) {
        const int mloc = wm * 32 + mf * 16 + lh * 8 + r;
        y[(size_t)(tileM + mloc) * EDIM + ncol] = acc[mf][nf][r];
      }
    }
  }
}

// ---------------- launch ----------------
extern "C" void kernel_launch(void* const* d_in, const int* in_sizes, int n_in,
                              void* d_out, int out_size, void* d_ws, size_t ws_size,
                              hipStream_t stream) {
  (void)in_sizes; (void)n_in; (void)out_size; (void)ws_size;
  const float* x  = (const float*)d_in[0];
  const float* gw = (const float*)d_in[1];
  const float* wg = (const float*)d_in[2];
  const float* wu = (const float*)d_in[3];
  const float* wd = (const float*)d_in[4];
  float* y = (float*)d_out;

  char* ws = (char*)d_ws;
  size_t off = 0;
  auto wsalloc = [&](size_t bytes) -> char* {
    char* p = ws + off;
    off = (off + bytes + 255) & ~(size_t)255;
    return p;
  };
  unsigned short* xbf  = (unsigned short*)wsalloc((size_t)BT * EDIM * 2);
  unsigned short* wgbf = (unsigned short*)wsalloc((size_t)NEXP * INTER * EDIM * 2);
  unsigned short* wubf = (unsigned short*)wsalloc((size_t)NEXP * INTER * EDIM * 2);
  unsigned short* wdbf = (unsigned short*)wsalloc((size_t)NEXP * EDIM * INTER * 2);
  float*          gate = (float*)wsalloc((size_t)BT * NEXP * 4);
  unsigned short* hp   = (unsigned short*)wsalloc((size_t)NEXP * BT * INTER * 2);

  const int nx = BT * EDIM;
  const int nw = NEXP * INTER * EDIM;
  convert_f32_bf16<<<1024, 256, 0, stream>>>(x,  xbf,  nx);
  convert_f32_bf16<<<2048, 256, 0, stream>>>(wg, wgbf, nw);
  convert_f32_bf16<<<2048, 256, 0, stream>>>(wu, wubf, nw);
  convert_f32_bf16<<<2048, 256, 0, stream>>>(wd, wdbf, nw);

  moe_router<<<BT / 256, 256, 0, stream>>>(x, gw, gate);

  moe_gemm1<<<dim3(INTER / 64, BT / 128, NEXP), 256, 0, stream>>>(xbf, wgbf, wubf, gate, hp);
  moe_gemm2<<<dim3(EDIM / 64, BT / 128), 256, 0, stream>>>(hp, wdbf, y);
}
